// HoloKVAttention_62672162784014
// MI455X (gfx1250) — matrix operand order repaired and checked by reading, not compiled, not run on hardware
//
#include <hip/hip_runtime.h>
#include <hip/hip_bf16.h>
#include <stdint.h>

#define HID   1024
#define NH    16
#define HD    64
#define KF    4
#define LR    64
#define BATCH 2
#define SLEN  2048
#define NS    (SLEN / KF)     // 512 slots
#define MTOK  (BATCH * SLEN)  // 4096 token rows

typedef __bf16 bf16_t;
typedef __attribute__((ext_vector_type(16))) __bf16 v16bf;
typedef __attribute__((ext_vector_type(8)))  float  v8f;
typedef int v4i_n __attribute__((vector_size(16)));   // matches builtin's param type

union AFrag { v16bf v; uint4 u[2]; };

// ---- CDNA5 async global->LDS copy (ASYNCcnt-tracked), with safe fallback ----
#if defined(__has_builtin)
# if __has_builtin(__builtin_amdgcn_global_load_async_to_lds_b128)
#  define HAVE_ASYNC_LDS 1
# endif
#endif
#ifndef HAVE_ASYNC_LDS
# define HAVE_ASYNC_LDS 0
#endif

#if defined(__has_builtin) && __has_builtin(__builtin_amdgcn_s_wait_asynccnt)
# define ASYNC_WAIT0() __builtin_amdgcn_s_wait_asynccnt(0)
#else
# define ASYNC_WAIT0() asm volatile("s_wait_asynccnt 0x0" ::: "memory")
#endif

__device__ __forceinline__ void async_copy16(const void* g, void* l) {
#if HAVE_ASYNC_LDS
  __builtin_amdgcn_global_load_async_to_lds_b128((v4i_n*)g, (v4i_n*)l, 0, 0);
#else
  *(uint4*)l = *(const uint4*)g;
#endif
}

__device__ __forceinline__ void async_fence_block() {
#if HAVE_ASYNC_LDS
  ASYNC_WAIT0();
#endif
  __syncthreads();
}

// ---------------------------------------------------------------- casts
__global__ void cast_f32_bf16(const float* __restrict__ src, bf16_t* __restrict__ dst, int n) {
  for (int i = blockIdx.x * blockDim.x + threadIdx.x; i < n; i += gridDim.x * blockDim.x)
    dst[i] = (bf16_t)src[i];
}

__global__ void fill_bf16_zero(bf16_t* __restrict__ dst, int n) {
  for (int i = blockIdx.x * blockDim.x + threadIdx.x; i < n; i += gridDim.x * blockDim.x)
    dst[i] = (bf16_t)0.0f;
}

// src[rows][cols] (f32, row-major) -> dst[cols][rows] (bf16)
__global__ void castT_f32_bf16(const float* __restrict__ src, bf16_t* __restrict__ dst,
                               int rows, int cols) {
  int n = rows * cols;
  for (int i = blockIdx.x * blockDim.x + threadIdx.x; i < n; i += gridDim.x * blockDim.x) {
    int r = i / cols, c = i % cols;
    dst[(size_t)c * rows + r] = (bf16_t)src[i];
  }
}

// ---------------------------------------------------------------- TN WMMA GEMM
// C[M,N] (f32) = A[M,K](bf16,row-major) @ Bt[N,K](bf16,row-major)^T   (+= if beta)
// M must be a multiple of 128; Bt must have >= round_up(N,128) valid (zero-padded) rows.
__launch_bounds__(256)
__global__ void gemm_tn_bf16(const bf16_t* __restrict__ A, const bf16_t* __restrict__ Bt,
                             float* __restrict__ C, int M, int N, int K, int beta) {
  __shared__ alignas(16) bf16_t As[128][40];
  __shared__ alignas(16) bf16_t Bs[128][40];
  const int tid  = threadIdx.x;
  const int lane = tid & 31, w = tid >> 5;
  const int wm = w & 3, wn = w >> 2;           // 4x2 wave grid -> 32x64 per wave
  const int m0 = blockIdx.y * 128, n0 = blockIdx.x * 128;
  const int lh = lane & 15;
  const int hi = (lane >= 16) ? 1 : 0;
  const int khalfA = hi ? 8 : 0;               // A-frag: k {0..7,16..23} / {8..15,24..31}
  const int khalfB = hi ? 8 : 0;               // B-frag: k {0..7,16..23} / {8..15,24..31}

  v8f zero = {};
  v8f acc[2][4];
  for (int i = 0; i < 2; i++)
    for (int j = 0; j < 4; j++) acc[i][j] = zero;

  // per-thread tile-load coordinates (straight-line, no guards)
  const int row0 = tid >> 2, cc0 = (tid & 3) * 8;
  const int row1 = (tid + 256) >> 2, cc1 = ((tid + 256) & 3) * 8;

  for (int kk = 0; kk < K; kk += 32) {
    async_copy16(A  + (size_t)(m0 + row0) * K + kk + cc0, &As[row0][cc0]);
    async_copy16(Bt + (size_t)(n0 + row0) * K + kk + cc0, &Bs[row0][cc0]);
    async_copy16(A  + (size_t)(m0 + row1) * K + kk + cc1, &As[row1][cc1]);
    async_copy16(Bt + (size_t)(n0 + row1) * K + kk + cc1, &Bs[row1][cc1]);
    if (kk + 32 < K) {
      __builtin_prefetch(A  + (size_t)(m0 + row0) * K + kk + 32 + cc0, 0, 3);
      __builtin_prefetch(Bt + (size_t)(n0 + row0) * K + kk + 32 + cc0, 0, 3);
    }
    async_fence_block();

    AFrag fa[2], fb[4];
    for (int fm = 0; fm < 2; fm++) {
      int r = wm * 32 + fm * 16 + lh;
      fa[fm].u[0] = *(const uint4*)&As[r][khalfA];
      fa[fm].u[1] = *(const uint4*)&As[r][khalfA + 16];
    }
    for (int fn = 0; fn < 4; fn++) {
      int r = wn * 64 + fn * 16 + lh;
      fb[fn].u[0] = *(const uint4*)&Bs[r][khalfB];
      fb[fn].u[1] = *(const uint4*)&Bs[r][khalfB + 16];
    }
    for (int fm = 0; fm < 2; fm++)
      for (int fn = 0; fn < 4; fn++)
        acc[fm][fn] = __builtin_amdgcn_wmma_f32_16x16x32_bf16(
            false, fa[fm].v, false, fb[fn].v, (short)0, acc[fm][fn], false, false);
    __syncthreads();
  }

  for (int fm = 0; fm < 2; fm++)
    for (int fn = 0; fn < 4; fn++)
      for (int r = 0; r < 8; r++) {
        int row = m0 + wm * 32 + fm * 16 + r + 8 * hi;
        int col = n0 + wn * 64 + fn * 16 + lh;
        if (col < N) {
          size_t idx = (size_t)row * N + col;
          float v = acc[fm][fn][r];
          C[idx] = beta ? (C[idx] + v) : v;
        }
      }
}

// ---------------------------------------------------------------- RoPE + CDMA compress
__launch_bounds__(64)
__global__ void rope_cdma_compress(const float* __restrict__ Q, const float* __restrict__ Kp,
                                   const float* __restrict__ Vp,
                                   float* __restrict__ cpV, bf16_t* __restrict__ aqb,
                                   float* __restrict__ cscore,
                                   float* __restrict__ fK, bf16_t* __restrict__ fKb,
                                   float* __restrict__ fV, bf16_t* __restrict__ fVtb) {
  int gid = blockIdx.x;
  int b = gid / (NH * NS);
  int rem = gid % (NH * NS);
  int h = rem / NS, slot = rem % NS;
  int t = threadIdx.x;
  __shared__ float red[64];

  int j = t & 31;
  float inv_freq = __expf(-(float)(2 * j) * (9.210340371976184f / 64.0f)); // 10000^(-2j/64)
  int p = (t + 32) & 63;
  float rsign = (t < 32) ? -1.0f : 1.0f;

  float runK = 0.f, runV = 0.f;
  for (int step = 0; step < KF; step++) {
    int s = slot * KF + step;
    size_t base = ((size_t)(b * SLEN + s)) * HID + h * HD;
    float fr = (float)s * inv_freq;
    float si, co;
    __sincosf(fr, &si, &co);

    float qv = Q[base + t],  qp = Q[base + p];
    float kv = Kp[base + t], kp = Kp[base + p];
    float vv = Vp[base + t];
    float qr = qv * co + rsign * qp * si;
    float kr = kv * co + rsign * kp * si;
    float ph = (__popc(step & (t & 3)) & 1) ? -1.0f : 1.0f;

    runK += kr * ph * 0.5f;
    runV += vv * ph * 0.5f;
    float aq = qr * ph;

    cpV[base + t] = runV;
    aqb[base + t] = (bf16_t)aq;

    red[t] = aq * runK;
    __syncthreads();
    for (int off = 32; off > 0; off >>= 1) {
      if (t < off) red[t] += red[t + off];
      __syncthreads();
    }
    if (t == 0) cscore[((size_t)(b * NH + h)) * SLEN + s] = red[0] * 0.125f;
    __syncthreads();
  }
  size_t kb = ((size_t)(b * NS + slot)) * HID + h * HD + t;
  fK[kb]  = runK;  fKb[kb] = (bf16_t)runK;
  fV[kb]  = runV;
  fVtb[((size_t)((b * NH + h) * HD + t)) * NS + slot] = (bf16_t)runV;
}

// ---------------------------------------------------------------- flash attention over slots
__launch_bounds__(128)
__global__ void holo_attention(const bf16_t* __restrict__ aqb, const bf16_t* __restrict__ fKb,
                               const bf16_t* __restrict__ fVtb, const float* __restrict__ cscore,
                               const float* __restrict__ cpV, const float* __restrict__ fV,
                               float* __restrict__ O, bf16_t* __restrict__ Ob) {
  __shared__ alignas(16) bf16_t Ks[32][72];
  __shared__ alignas(16) bf16_t Vt[64][40];
  __shared__ alignas(16) bf16_t Ps[4][16][40];

  const int tid = threadIdx.x, lane = tid & 31, wave = tid >> 5;
  const int bh = blockIdx.y, b = bh / NH, h = bh % NH;
  const int q0wg = blockIdx.x * 64;
  const int q0 = q0wg + wave * 16;
  const int lh = lane & 15, hi = (lane >= 16) ? 1 : 0;
  const int khalfA = hi ? 8 : 0;
  const int khalfB = hi ? 8 : 0;

  AFrag aq[2];
  {
    size_t base = ((size_t)(b * SLEN + q0 + lh)) * HID + h * HD;
    for (int c = 0; c < 2; c++) {
      aq[c].u[0] = *(const uint4*)(aqb + base + c * 32 + khalfA);
      aq[c].u[1] = *(const uint4*)(aqb + base + c * 32 + khalfA + 16);
    }
  }
  float cs[8], mrow[8], lrow[8];
  for (int r = 0; r < 8; r++) {
    int q = q0 + r + 8 * hi;
    cs[r] = cscore[((size_t)(b * NH + h)) * SLEN + q];
    mrow[r] = -1e30f;
    lrow[r] = 0.f;
  }
  v8f zero = {};
  v8f acc[4];
  for (int i = 0; i < 4; i++) acc[i] = zero;

  // per-thread staging coordinates
  const int kr0 = tid >> 3,         kc0 = (tid & 7) * 8;          // K tile 32x64
  const int kr1 = (tid + 128) >> 3, kc1 = ((tid + 128) & 7) * 8;
  const int vr0 = tid >> 2,         vc0 = (tid & 3) * 8;          // Vt tile 64x32
  const int vr1 = (tid + 128) >> 2, vc1 = ((tid + 128) & 3) * 8;

  const int maxslot = (q0wg + 63) >> 2;
  const int nblk = maxslot / 32 + 1;

  for (int jb = 0; jb < nblk; jb++) {
    __syncthreads();
    async_copy16(fKb  + ((size_t)(b * NS + jb * 32 + kr0)) * HID + h * HD + kc0, &Ks[kr0][kc0]);
    async_copy16(fKb  + ((size_t)(b * NS + jb * 32 + kr1)) * HID + h * HD + kc1, &Ks[kr1][kc1]);
    async_copy16(fVtb + ((size_t)((b * NH + h) * HD + vr0)) * NS + jb * 32 + vc0, &Vt[vr0][vc0]);
    async_copy16(fVtb + ((size_t)((b * NH + h) * HD + vr1)) * NS + jb * 32 + vc1, &Vt[vr1][vc1]);
    async_fence_block();

    float pbuf[2][8];
    float mx[8];
    for (int r = 0; r < 8; r++) mx[r] = -3e38f;
    for (int jj = 0; jj < 2; jj++) {
      AFrag fbc[2];
      int krow = jj * 16 + lh;
      for (int c = 0; c < 2; c++) {
        fbc[c].u[0] = *(const uint4*)&Ks[krow][c * 32 + khalfB];
        fbc[c].u[1] = *(const uint4*)&Ks[krow][c * 32 + khalfB + 16];
      }
      v8f sc = zero;
      sc = __builtin_amdgcn_wmma_f32_16x16x32_bf16(false, aq[0].v, false, fbc[0].v, (short)0, sc, false, false);
      sc = __builtin_amdgcn_wmma_f32_16x16x32_bf16(false, aq[1].v, false, fbc[1].v, (short)0, sc, false, false);
      int slotc = jb * 32 + jj * 16 + lh;
      for (int r = 0; r < 8; r++) {
        int q = q0 + r + 8 * hi;
        int cslot = q >> 2;
        float s = sc[r] * 0.125f;
        if (slotc > cslot)       s = -3e38f;
        else if (slotc == cslot) s = cs[r];
        pbuf[jj][r] = s;
        mx[r] = fmaxf(mx[r], s);
      }
    }
    for (int r = 0; r < 8; r++) {
      float v = mx[r];
      for (int msk = 1; msk <= 8; msk <<= 1) v = fmaxf(v, __shfl_xor(v, msk, 32));
      float mnew = fmaxf(mrow[r], v);
      float fac = __expf(mrow[r] - mnew);
      mrow[r] = mnew;
      lrow[r] *= fac;
      for (int i = 0; i < 4; i++) acc[i][r] *= fac;
      float rs = 0.f;
      for (int jj = 0; jj < 2; jj++) {
        float pe = __expf(pbuf[jj][r] - mnew);
        pbuf[jj][r] = pe;
        rs += pe;
      }
      for (int msk = 1; msk <= 8; msk <<= 1) rs += __shfl_xor(rs, msk, 32);
      lrow[r] += rs;
    }
    for (int jj = 0; jj < 2; jj++)
      for (int r = 0; r < 8; r++)
        Ps[wave][r + 8 * hi][jj * 16 + lh] = (bf16_t)pbuf[jj][r];
    __syncthreads();

    AFrag pa;
    pa.u[0] = *(const uint4*)&Ps[wave][lh][khalfA];
    pa.u[1] = *(const uint4*)&Ps[wave][lh][khalfA + 16];
    for (int jj2 = 0; jj2 < 4; jj2++) {
      AFrag bv;
      int vr = jj2 * 16 + lh;
      bv.u[0] = *(const uint4*)&Vt[vr][khalfB];
      bv.u[1] = *(const uint4*)&Vt[vr][khalfB + 16];
      acc[jj2] = __builtin_amdgcn_wmma_f32_16x16x32_bf16(
          false, pa.v, false, bv.v, (short)0, acc[jj2], false, false);
    }
  }

  for (int jj2 = 0; jj2 < 4; jj2++)
    for (int r = 0; r < 8; r++) {
      int q = q0 + r + 8 * hi;
      int dl = jj2 * 16 + lh;
      size_t idx = ((size_t)(b * SLEN + q)) * HID + h * HD + dl;
      float invl = 1.0f / lrow[r];
      float wcur = __expf(cs[r] - mrow[r]) * invl;
      float corr = cpV[idx] - fV[((size_t)(b * NS + (q >> 2))) * HID + h * HD + dl];
      float outv = acc[jj2][r] * invl + wcur * corr;
      float ph = (__popc((q & 3) & (dl & 3)) & 1) ? -1.0f : 1.0f;
      outv *= ph * 2.0f;
      O[idx] = outv;
      Ob[idx] = (bf16_t)outv;
    }
}

// ---------------------------------------------------------------- launcher
extern "C" void kernel_launch(void* const* d_in, const int* in_sizes, int n_in,
                              void* d_out, int out_size, void* d_ws, size_t ws_size,
                              hipStream_t stream) {
  (void)in_sizes; (void)n_in; (void)out_size; (void)ws_size;
  const float* X  = (const float*)d_in[0];
  const float* Wq = (const float*)d_in[1];
  const float* Wk = (const float*)d_in[2];
  const float* Wv = (const float*)d_in[3];
  const float* Wo = (const float*)d_in[4];
  const float* Aq = (const float*)d_in[5];
  const float* Bq = (const float*)d_in[6];
  const float* Av = (const float*)d_in[7];
  const float* Bv = (const float*)d_in[8];
  const float* Ao = (const float*)d_in[9];
  const float* Bo = (const float*)d_in[10];
  float* Y = (float*)d_out;

  size_t off = 0;
  auto alloc = [&](size_t bytes) -> char* {
    char* p = (char*)d_ws + off;
    off += (bytes + 255) & ~(size_t)255;
    return p;
  };
  bf16_t* Xb  = (bf16_t*)alloc((size_t)MTOK * HID * 2);
  bf16_t* WqT = (bf16_t*)alloc((size_t)HID * HID * 2);
  bf16_t* WkT = (bf16_t*)alloc((size_t)HID * HID * 2);
  bf16_t* WvT = (bf16_t*)alloc((size_t)HID * HID * 2);
  bf16_t* WoT = (bf16_t*)alloc((size_t)HID * HID * 2);
  bf16_t* AqT = (bf16_t*)alloc((size_t)128 * HID * 2);   // 64 real rows + 64 zero-pad
  bf16_t* AvT = (bf16_t*)alloc((size_t)128 * HID * 2);
  bf16_t* AoT = (bf16_t*)alloc((size_t)128 * HID * 2);
  bf16_t* BqT = (bf16_t*)alloc((size_t)HID * LR * 2);
  bf16_t* BvT = (bf16_t*)alloc((size_t)HID * LR * 2);
  bf16_t* BoT = (bf16_t*)alloc((size_t)HID * LR * 2);
  float*  Qf  = (float*)alloc((size_t)MTOK * HID * 4);
  float*  Kf  = (float*)alloc((size_t)MTOK * HID * 4);
  float*  Vf  = (float*)alloc((size_t)MTOK * HID * 4);
  float*  t1  = (float*)alloc((size_t)MTOK * LR * 4);
  bf16_t* t1b = (bf16_t*)alloc((size_t)MTOK * LR * 2);
  bf16_t* aqb = (bf16_t*)alloc((size_t)MTOK * HID * 2);
  float*  cpV = (float*)alloc((size_t)MTOK * HID * 4);
  float*  csc = (float*)alloc((size_t)BATCH * NH * SLEN * 4);
  float*  fK  = (float*)alloc((size_t)BATCH * NS * HID * 4);
  bf16_t* fKb = (bf16_t*)alloc((size_t)BATCH * NS * HID * 2);
  float*  fV  = (float*)alloc((size_t)BATCH * NS * HID * 4);
  bf16_t* fVt = (bf16_t*)alloc((size_t)BATCH * NS * HID * 2);
  float*  O   = (float*)alloc((size_t)MTOK * HID * 4);
  bf16_t* Obf = (bf16_t*)alloc((size_t)MTOK * HID * 2);

  // 1) casts / transposes (+ zero-pad rows 64..127 of the transposed LoRA-A mats)
  cast_f32_bf16<<<4096, 256, 0, stream>>>(X, Xb, MTOK * HID);
  castT_f32_bf16<<<2048, 256, 0, stream>>>(Wq, WqT, HID, HID);
  castT_f32_bf16<<<2048, 256, 0, stream>>>(Wk, WkT, HID, HID);
  castT_f32_bf16<<<2048, 256, 0, stream>>>(Wv, WvT, HID, HID);
  castT_f32_bf16<<<2048, 256, 0, stream>>>(Wo, WoT, HID, HID);
  castT_f32_bf16<<<256, 256, 0, stream>>>(Aq, AqT, HID, LR);
  castT_f32_bf16<<<256, 256, 0, stream>>>(Av, AvT, HID, LR);
  castT_f32_bf16<<<256, 256, 0, stream>>>(Ao, AoT, HID, LR);
  fill_bf16_zero<<<64, 256, 0, stream>>>(AqT + (size_t)LR * HID, (128 - LR) * HID);
  fill_bf16_zero<<<64, 256, 0, stream>>>(AvT + (size_t)LR * HID, (128 - LR) * HID);
  fill_bf16_zero<<<64, 256, 0, stream>>>(AoT + (size_t)LR * HID, (128 - LR) * HID);
  castT_f32_bf16<<<256, 256, 0, stream>>>(Bq, BqT, LR, HID);
  castT_f32_bf16<<<256, 256, 0, stream>>>(Bv, BvT, LR, HID);
  castT_f32_bf16<<<256, 256, 0, stream>>>(Bo, BoT, LR, HID);

  dim3 gFull(HID / 128, MTOK / 128);   // N=1024
  dim3 gLora(1, MTOK / 128);           // N=64 (Bt zero-padded to 128 rows)
  // 2) projections
  gemm_tn_bf16<<<gFull, 256, 0, stream>>>(Xb, WqT, Qf, MTOK, HID, HID, 0);
  gemm_tn_bf16<<<gFull, 256, 0, stream>>>(Xb, WkT, Kf, MTOK, HID, HID, 0);
  gemm_tn_bf16<<<gFull, 256, 0, stream>>>(Xb, WvT, Vf, MTOK, HID, HID, 0);
  // LoRA q
  gemm_tn_bf16<<<gLora, 256, 0, stream>>>(Xb, AqT, t1, MTOK, LR, HID, 0);
  cast_f32_bf16<<<1024, 256, 0, stream>>>(t1, t1b, MTOK * LR);
  gemm_tn_bf16<<<gFull, 256, 0, stream>>>(t1b, BqT, Qf, MTOK, HID, LR, 1);
  // LoRA v
  gemm_tn_bf16<<<gLora, 256, 0, stream>>>(Xb, AvT, t1, MTOK, LR, HID, 0);
  cast_f32_bf16<<<1024, 256, 0, stream>>>(t1, t1b, MTOK * LR);
  gemm_tn_bf16<<<gFull, 256, 0, stream>>>(t1b, BvT, Vf, MTOK, HID, LR, 1);

  // 3) RoPE + CDMA compression
  rope_cdma_compress<<<BATCH * NH * NS, 64, 0, stream>>>(Qf, Kf, Vf, cpV, aqb, csc,
                                                         fK, fKb, fV, fVt);
  // 4) attention
  holo_attention<<<dim3(SLEN / 64, BATCH * NH), 128, 0, stream>>>(aqb, fKb, fVt, csc,
                                                                  cpV, fV, O, Obf);
  // 5) output projection + LoRA o
  gemm_tn_bf16<<<gFull, 256, 0, stream>>>(Obf, WoT, Y, MTOK, HID, HID, 0);
  gemm_tn_bf16<<<gLora, 256, 0, stream>>>(Obf, AoT, t1, MTOK, LR, HID, 0);
  cast_f32_bf16<<<1024, 256, 0, stream>>>(t1, t1b, MTOK * LR);
  gemm_tn_bf16<<<gFull, 256, 0, stream>>>(t1b, BoT, Y, MTOK, HID, LR, 1);
}
